// AttnBlock_21663815041727
// MI455X (gfx1250) — matrix operand order repaired and checked by reading, not compiled, not run on hardware
//
#include <hip/hip_runtime.h>

// ---------------------------------------------------------------------------
// AttnBlock for MI455X (gfx1250): bf16 WMMA everywhere, flash-attention core
// with TDM (tensor_load_to_lds) + async-to-LDS double-buffered K/V staging.
// ---------------------------------------------------------------------------

typedef __attribute__((ext_vector_type(16))) __bf16 v16bf;
typedef __attribute__((ext_vector_type(8)))  float  v8f;
typedef unsigned __attribute__((ext_vector_type(4))) uint32x4;
typedef unsigned __attribute__((ext_vector_type(8))) uint32x8;

#define B_   16
#define C_   512
#define HW_  4096
#define G_   32
#define CG_  16          // C_/G_
#define KT_  64          // key tile for flash attention
#define SCALE_ 0.04419417382415922f   // 1/sqrt(512)

__device__ __forceinline__ v8f wmma_bf16(v16bf a, v16bf b, v8f c) {
  return __builtin_amdgcn_wmma_f32_16x16x32_bf16(
      false, a, false, b, (short)0, c, false, false);
}

// A-frag (16x32 bf16): row-major source, element (m,k) at base[m*ld + k].
__device__ __forceinline__ v16bf load_fragA(const __bf16* base, int ld, int lane) {
  const int r = lane & 15, h = lane >> 4;
  const __bf16* p = base + (size_t)r * ld + h * 16;
  v16bf f;
#pragma unroll
  for (int e = 0; e < 8; ++e) f[e] = p[e];
#pragma unroll
  for (int e = 0; e < 8; ++e) f[e + 8] = p[8 + e];
  return f;
}

// B-frag (32x16 bf16) from an N-major source: element (k,n) = base[n*ld + k].
__device__ __forceinline__ v16bf load_fragB_nk(const __bf16* base, int ld, int lane) {
  const int n = lane & 15, h = lane >> 4;
  const __bf16* p = base + (size_t)n * ld + h * 16;
  v16bf f;
#pragma unroll
  for (int e = 0; e < 16; ++e) f[e] = p[e];
  return f;
}

// B-frag (32x16 bf16) from a K-major source: element (k,n) = base[k*ld + n].
__device__ __forceinline__ v16bf load_fragB_kn(const __bf16* base, int ld, int lane) {
  const int n = lane & 15, h = lane >> 4;
  const __bf16* p = base + (size_t)(h * 16) * ld + n;
  v16bf f;
#pragma unroll
  for (int e = 0; e < 16; ++e) f[e] = p[(size_t)e * ld];
  return f;
}

// ---------------------------------------------------------------------------
// CDNA5 async/tensor data movers (inline asm; assembler-verified round 2).
// ---------------------------------------------------------------------------
__device__ __forceinline__ void async_copy_b128(const void* gsrc, void* lds_dst) {
  unsigned lds = (unsigned)(size_t)lds_dst;           // low 32 bits = LDS offset
  unsigned long long ga = (unsigned long long)(size_t)gsrc;
  asm volatile("global_load_async_to_lds_b128 %0, %1, off"
               :: "v"(lds), "v"(ga) : "memory");
}

// TDM: contiguous 1-D tile of `nelem` bf16 elements, global -> LDS.
__device__ __forceinline__ void tdm_load_1d(const void* gsrc, void* lds_dst,
                                            unsigned nelem) {
  unsigned long long ga = (unsigned long long)(size_t)gsrc;
  uint32x4 g0;
  g0[0] = 1u;                                   // count=1, user mode
  g0[1] = (unsigned)(size_t)lds_dst;            // lds_addr
  g0[2] = (unsigned)ga;                         // global_addr[31:0]
  g0[3] = (unsigned)(ga >> 32) | (2u << 30);    // global_addr[56:32] | type=2
  uint32x8 g1;
  g1[0] = 1u << 16;                             // wg_mask=0, data_size=2 bytes
  g1[1] = (nelem & 0xFFFFu) << 16;              // tensor_dim0[15:0]
  g1[2] = (nelem >> 16) | (1u << 16);           // tensor_dim0[31:16], tensor_dim1=1
  g1[3] = (nelem & 0xFFFFu) << 16;              // tile_dim0
  g1[4] = 1u;                                   // tile_dim1=1, tile_dim2=0
  g1[5] = nelem;                                // tensor_dim0_stride[31:0]
  g1[6] = 0u;
  g1[7] = 0u;
  asm volatile("tensor_load_to_lds %0, %1" :: "s"(g0), "s"(g1) : "memory");
}

// ---------------------------------------------------------------------------
// Kernel 0: convert the four 512x512 fp32 weight matrices to bf16.
// ---------------------------------------------------------------------------
__global__ __launch_bounds__(256) void cvt_w_kernel(
    const float* __restrict__ wq, const float* __restrict__ wk,
    const float* __restrict__ wv, const float* __restrict__ wo,
    __bf16* __restrict__ out) {
  const float* srcs[4] = {wq, wk, wv, wo};
  int i = blockIdx.x * 256 + threadIdx.x;            // 4 * 2^18 total
  out[i] = (__bf16)srcs[i >> 18][i & 262143];
}

// ---------------------------------------------------------------------------
// Kernel 1: GroupNorm -> bf16 token-major hn[b, t, c].
// ---------------------------------------------------------------------------
__global__ __launch_bounds__(256) void gn_kernel(
    const float* __restrict__ x, const float* __restrict__ gw,
    const float* __restrict__ gb, __bf16* __restrict__ hn) {
  const int bg = blockIdx.x;
  const int bat = bg / G_, g = bg % G_;
  const float* xp = x + ((size_t)bat * C_ + (size_t)g * CG_) * HW_;
  const int tid = threadIdx.x;
  const int N = CG_ * HW_;                           // 65536

  float s = 0.f, ss = 0.f;
  for (int i = tid; i < N; i += 256) { float v = xp[i]; s += v; ss += v * v; }

  __shared__ float red0[256], red1[256];
  red0[tid] = s; red1[tid] = ss;
  __syncthreads();
  for (int off = 128; off > 0; off >>= 1) {
    if (tid < off) { red0[tid] += red0[tid + off]; red1[tid] += red1[tid + off]; }
    __syncthreads();
  }
  const float mean = red0[0] * (1.f / N);
  const float var  = red1[0] * (1.f / N) - mean * mean;
  const float rstd = rsqrtf(var + 1e-5f);

  for (int i = tid; i < N; i += 256) {
    int ci = i >> 12;
    int t  = i & (HW_ - 1);
    int c  = g * CG_ + ci;
    float v = (xp[i] - mean) * rstd * gw[c] + gb[c];
    hn[((size_t)bat * HW_ + t) * C_ + c] = (__bf16)v;
  }
}

// ---------------------------------------------------------------------------
// Kernel 2: QKV projection. Y = Hn @ W^T + bias for op in {q,k,v}.
// Block 256 = 8 waves; each wave computes a 16x64 tile (4 N-frags reuse one
// A-frag per k-step -> 4 WMMAs per 10 loads, 4x less A re-read).
// grid (M/128, C/64, 3).
// ---------------------------------------------------------------------------
__global__ __launch_bounds__(256) void qkv_gemm_kernel(
    const __bf16* __restrict__ hn, const __bf16* __restrict__ wbf,
    const float* __restrict__ bq, const float* __restrict__ bk,
    const float* __restrict__ bv,
    __bf16* __restrict__ qo, __bf16* __restrict__ ko, __bf16* __restrict__ vo) {
  const int op = blockIdx.z;
  const __bf16* W = wbf + (size_t)op * C_ * C_;      // [N=co, K=c] row-major
  const float* bias = (op == 0) ? bq : (op == 1) ? bk : bv;
  __bf16* out = (op == 0) ? qo : (op == 1) ? ko : vo;

  const int n0   = blockIdx.y * 64;
  const int wave = threadIdx.x >> 5;
  const int lane = threadIdx.x & 31;
  const int m0   = blockIdx.x * 128 + wave * 16;

  __shared__ __bf16 wt[64 * C_];                     // 64 KB weight tile
  {
    const uint4* src = (const uint4*)(W + (size_t)n0 * C_);
    uint4* dst = (uint4*)wt;
    for (int i = threadIdx.x; i < 64 * C_ / 8; i += 256) dst[i] = src[i];
  }
  __syncthreads();

  const int n = lane & 15, h = lane >> 4;
  v8f acc[4];
#pragma unroll
  for (int f = 0; f < 4; ++f) {
    const float bsc = bias[n0 + f * 16 + n];
#pragma unroll
    for (int j = 0; j < 8; ++j) acc[f][j] = bsc;
  }

  for (int kk = 0; kk < C_; kk += 32) {
    v16bf a = load_fragA(hn + (size_t)m0 * C_ + kk, C_, lane);
#pragma unroll
    for (int f = 0; f < 4; ++f) {
      v16bf bb = load_fragB_nk(wt + (size_t)(f * 16) * C_ + kk, C_, lane);
      acc[f] = wmma_bf16(a, bb, acc[f]);
    }
  }
#pragma unroll
  for (int f = 0; f < 4; ++f)
#pragma unroll
    for (int j = 0; j < 8; ++j)
      out[(size_t)(m0 + j + 8 * h) * C_ + n0 + f * 16 + n] = (__bf16)acc[f][j];
}

// ---------------------------------------------------------------------------
// Kernel 3: flash attention. Block 256 = 8 waves = 2 query-groups of 4 waves;
// 32 queries per workgroup share the double-buffered 64-key K/V LDS tiles
// (halves K/V L2 traffic vs 16q/WG). K via TDM (8 descriptors, one 8-row
// slice per wave), V via per-lane async b128; DMA of tile i+1 overlaps WMMA
// of tile i; sync via TENSORcnt/ASYNCcnt partial waits.
// LDS: 2*(64+64)K K/V + 2*16K Q + P/stats ~= 293 KB (320 KB CDNA5 WGP LDS).
// ---------------------------------------------------------------------------
__global__ __launch_bounds__(256) void attn_kernel(
    const __bf16* __restrict__ q, const __bf16* __restrict__ k,
    const __bf16* __restrict__ v, __bf16* __restrict__ o) {
  const int bat  = blockIdx.y;
  const int q0   = blockIdx.x * 32;                  // 32 queries per block
  const int w8   = threadIdx.x >> 5;                 // wave 0..7
  const int grp  = threadIdx.x >> 7;                 // query group 0..1
  const int wv   = w8 & 3;                           // wave-in-group 0..3
  const int lane = threadIdx.x & 31;
  const int h = lane >> 4, n = lane & 15;

  const __bf16* Qp = q + ((size_t)bat * HW_ + q0 + grp * 16) * C_;
  const __bf16* Kp = k + (size_t)bat * HW_ * C_;
  const __bf16* Vp = v + (size_t)bat * HW_ * C_;

  __shared__ __bf16 qs[2][16 * C_];                  // 32 KB
  __shared__ __bf16 ks[2][KT_ * C_];                 // 128 KB (double buffer)
  __shared__ __bf16 vs[2][KT_ * C_];                 // 128 KB (double buffer)
  __shared__ __bf16 ps[2][16 * KT_];                 // 4 KB (P tiles)
  __shared__ float m_run[2][16], l_run[2][16];
  __shared__ float wmax[2][4][16], wsum[2][4][16];

  {
    const uint4* src = (const uint4*)Qp;
    uint4* dst = (uint4*)&qs[grp][0];
    for (int i = threadIdx.x & 127; i < 16 * C_ / 8; i += 128) dst[i] = src[i];
  }
  if (threadIdx.x < 32) {
    m_run[threadIdx.x >> 4][threadIdx.x & 15] = -1e30f;
    l_run[threadIdx.x >> 4][threadIdx.x & 15] = 0.f;
  }

  v8f o_acc[8];                                      // wave owns C cols [wv*128,+128)
#pragma unroll
  for (int f = 0; f < 8; ++f) o_acc[f] = (v8f){};

  // ---- issue DMA for one K/V tile into buffer `buf` ----
  auto issue_tile = [&](int kt0, int buf) {
    // K tile: each of the 8 waves TDM-loads an 8-row slice (8*512 bf16).
    tdm_load_1d(Kp + ((size_t)kt0 + w8 * 8) * C_,
                &ks[buf][(size_t)(w8 * 8) * C_], 8u * C_);
    // V tile: 4096 x b128 async copies spread over 256 lanes (16 per lane).
    const char* gsrc = (const char*)(Vp + (size_t)kt0 * C_);
    char* ldst = (char*)&vs[buf][0];
    for (int i = threadIdx.x; i < KT_ * C_ / 8; i += 256)
      async_copy_b128(gsrc + (size_t)i * 16, ldst + (size_t)i * 16);
  };

  issue_tile(0, 0);                                  // preload first tile
  __syncthreads();                                   // (covers m_run/l_run init)

  const int NT = HW_ / KT_;                          // 64 tiles
  for (int ti = 0; ti < NT; ++ti) {
    const int buf = ti & 1;
    const bool more = (ti + 1 < NT);
    if (more) issue_tile((ti + 1) * KT_, buf ^ 1);   // overlap DMA with compute

    // Wait for *current* tile only (in-order per-wave counters):
    // 1 TDM + 16 async ops may stay outstanding for the next tile.
    if (more) {
      __builtin_amdgcn_s_wait_tensorcnt((short)1);
      asm volatile("s_wait_asynccnt 0x10" ::: "memory");
    } else {
      __builtin_amdgcn_s_wait_tensorcnt((short)0);
      asm volatile("s_wait_asynccnt 0x0" ::: "memory");
    }
    __syncthreads();                                 // tile visible to all waves

    const __bf16* ksb = &ks[buf][0];
    const __bf16* vsb = &vs[buf][0];

    // ---- S = scale * Q K^T : wave computes 16 queries x 16 keys ----
    v8f s = {};
    for (int kk = 0; kk < C_; kk += 32) {
      v16bf a  = load_fragA(&qs[grp][0] + kk, C_, lane);
      v16bf bb = load_fragB_nk(ksb + (size_t)(wv * 16) * C_ + kk, C_, lane);
      s = wmma_bf16(a, bb, s);
    }

    // ---- per-row max inside wave frag ----
    float rmax8[8], mold8[8], mnew8[8], rsum8[8];
#pragma unroll
    for (int j = 0; j < 8; ++j) {
      float mval = s[j] * SCALE_;
      s[j] = mval;
      for (int d = 1; d < 16; d <<= 1) mval = fmaxf(mval, __shfl_xor(mval, d, 32));
      rmax8[j] = mval;
    }
    if (n == 0) {
#pragma unroll
      for (int j = 0; j < 8; ++j) wmax[grp][wv][j + 8 * h] = rmax8[j];
    }
    __syncthreads();

    // ---- new row max across group's waves, P = exp(S - mnew), row sums ----
#pragma unroll
    for (int j = 0; j < 8; ++j) {
      const int r = j + 8 * h;
      float mo = m_run[grp][r], mn = mo;
#pragma unroll
      for (int w = 0; w < 4; ++w) mn = fmaxf(mn, wmax[grp][w][r]);
      mold8[j] = mo; mnew8[j] = mn;
      float p = __expf(s[j] - mn);
      s[j] = p;
      float sv2 = p;
      for (int d = 1; d < 16; d <<= 1) sv2 += __shfl_xor(sv2, d, 32);
      rsum8[j] = sv2;
      ps[grp][r * KT_ + wv * 16 + n] = (__bf16)p;    // restage P for A-frag use
    }
    if (n == 0) {
#pragma unroll
      for (int j = 0; j < 8; ++j) wsum[grp][wv][j + 8 * h] = rsum8[j];
    }
    __syncthreads();

    // ---- update running stats (16 lanes per group) ----
    if ((threadIdx.x & 127) < 16) {
      const int gg = threadIdx.x >> 7;
      const int r = threadIdx.x & 15;
      float mo = m_run[gg][r], mn = mo;
      for (int w = 0; w < 4; ++w) mn = fmaxf(mn, wmax[gg][w][r]);
      float corr = __expf(mo - mn);
      float ls = l_run[gg][r] * corr;
      for (int w = 0; w < 4; ++w) ls += wsum[gg][w][r];
      m_run[gg][r] = mn; l_run[gg][r] = ls;
    }

    // ---- rescale O, then O += P V ----
    float corr8[8];
#pragma unroll
    for (int j = 0; j < 8; ++j) corr8[j] = __expf(mold8[j] - mnew8[j]);
#pragma unroll
    for (int f = 0; f < 8; ++f)
#pragma unroll
      for (int j = 0; j < 8; ++j) o_acc[f][j] *= corr8[j];

    for (int t = 0; t < KT_; t += 32) {
      v16bf a = load_fragA(&ps[grp][0] + t, KT_, lane);
#pragma unroll
      for (int f = 0; f < 8; ++f) {
        v16bf bb = load_fragB_kn(vsb + (size_t)t * C_ + wv * 128 + f * 16, C_, lane);
        o_acc[f] = wmma_bf16(a, bb, o_acc[f]);
      }
    }
    __syncthreads();                                 // compute done before reuse
  }

  // ---- epilogue: O /= l_run, store bf16 token-major ----
#pragma unroll
  for (int f = 0; f < 8; ++f) {
    const int c0 = wv * 128 + f * 16;
#pragma unroll
    for (int j = 0; j < 8; ++j) {
      const int r = j + 8 * h;
      float val = o_acc[f][j] / l_run[grp][r];
      o[((size_t)bat * HW_ + q0 + grp * 16 + r) * C_ + c0 + n] = (__bf16)val;
    }
  }
}

// ---------------------------------------------------------------------------
// Kernel 4: output projection + residual, transposed store to [B, C, H, W].
// Same 16x64-per-wave blocking as kernel 2.
// ---------------------------------------------------------------------------
__global__ __launch_bounds__(256) void proj_kernel(
    const __bf16* __restrict__ attn, const __bf16* __restrict__ wo_bf,
    const float* __restrict__ bo, const float* __restrict__ x,
    float* __restrict__ out) {
  const int n0   = blockIdx.y * 64;
  const int wave = threadIdx.x >> 5;
  const int lane = threadIdx.x & 31;
  const int m0   = blockIdx.x * 128 + wave * 16;

  __shared__ __bf16 wt[64 * C_];
  {
    const uint4* src = (const uint4*)(wo_bf + (size_t)n0 * C_);
    uint4* dst = (uint4*)wt;
    for (int i = threadIdx.x; i < 64 * C_ / 8; i += 256) dst[i] = src[i];
  }
  __syncthreads();

  const int n = lane & 15, h = lane >> 4;
  v8f acc[4];
#pragma unroll
  for (int f = 0; f < 4; ++f) {
    const float bsc = bo[n0 + f * 16 + n];
#pragma unroll
    for (int j = 0; j < 8; ++j) acc[f][j] = bsc;
  }

  for (int kk = 0; kk < C_; kk += 32) {
    v16bf a = load_fragA(attn + (size_t)m0 * C_ + kk, C_, lane);
#pragma unroll
    for (int f = 0; f < 4; ++f) {
      v16bf bb = load_fragB_nk(wt + (size_t)(f * 16) * C_ + kk, C_, lane);
      acc[f] = wmma_bf16(a, bb, acc[f]);
    }
  }
#pragma unroll
  for (int f = 0; f < 4; ++f)
#pragma unroll
    for (int j = 0; j < 8; ++j) {
      const int gm = m0 + j + 8 * h;                 // global token
      const int bat = gm >> 12, t = gm & (HW_ - 1);
      const size_t idx = ((size_t)bat * C_ + n0 + f * 16 + n) * HW_ + t;
      out[idx] = x[idx] + acc[f][j];
    }
}

// ---------------------------------------------------------------------------
extern "C" void kernel_launch(void* const* d_in, const int* in_sizes, int n_in,
                              void* d_out, int out_size, void* d_ws, size_t ws_size,
                              hipStream_t stream) {
  const float* x  = (const float*)d_in[0];
  const float* gw = (const float*)d_in[1];
  const float* gb = (const float*)d_in[2];
  const float* wq = (const float*)d_in[3];
  const float* bq = (const float*)d_in[4];
  const float* wk = (const float*)d_in[5];
  const float* bk = (const float*)d_in[6];
  const float* wv = (const float*)d_in[7];
  const float* bv = (const float*)d_in[8];
  const float* wo = (const float*)d_in[9];
  const float* bo = (const float*)d_in[10];
  float* out = (float*)d_out;

  char* ws = (char*)d_ws;
  const size_t act = (size_t)B_ * HW_ * C_ * sizeof(__bf16);   // 64 MB
  size_t off = 0;
  __bf16* wbf  = (__bf16*)(ws + off); off += (size_t)4 * C_ * C_ * sizeof(__bf16);
  __bf16* hn   = (__bf16*)(ws + off); off += act;
  __bf16* qb   = (__bf16*)(ws + off); off += act;
  __bf16* kb   = (__bf16*)(ws + off); off += act;
  __bf16* vb   = (__bf16*)(ws + off); off += act;
  __bf16* attn = (__bf16*)(ws + off); off += act;

  cvt_w_kernel<<<4096, 256, 0, stream>>>(wq, wk, wv, wo, wbf);
  gn_kernel<<<B_ * G_, 256, 0, stream>>>(x, gw, gb, hn);
  qkv_gemm_kernel<<<dim3((B_ * HW_) / 128, C_ / 64, 3), 256, 0, stream>>>(
      hn, wbf, bq, bk, bv, qb, kb, vb);
  attn_kernel<<<dim3(HW_ / 32, B_), 256, 0, stream>>>(qb, kb, vb, attn);
  proj_kernel<<<dim3((B_ * HW_) / 128, C_ / 64), 256, 0, stream>>>(
      attn, wbf + (size_t)3 * C_ * C_, bo, x, out);
}
